// BranchedPolicyNetwork_1829656068175
// MI455X (gfx1250) — matrix operand order repaired and checked by reading, not compiled, not run on hardware
//
#include <hip/hip_runtime.h>
#include <hip/hip_bf16.h>
#include <math.h>

// CDNA5 / gfx1250: wave32, WMMA 16x16x32 f16 -> f32 accum.

typedef _Float16 v16h __attribute__((ext_vector_type(16)));
typedef _Float16 h8   __attribute__((ext_vector_type(8)));
typedef float    v8f  __attribute__((ext_vector_type(8)));

#define B_ROWS 32768
#define A_BR   64
#define L_DIM  1024
#define N_DIM  128            // 2*A
#define KCHUNK 32             // K per WMMA
#define NKC    (L_DIM / KCHUNK)   // 32 K-chunks
#define MBLK   128            // rows per workgroup (8 m-tiles x 16 rows, all per wave)
#define LSTR   40             // LDS row stride in halfs (32 data + 8 pad, 80B rows)

// ---------------------------------------------------------------------------
// Prologue: pack W [128 x 1024] f32 row-major into per-lane f16 B-fragments.
// Layout: Wp[((ntile*NKC + kc)*32 + lane)*16 + j]
//   lane: n = ntile*16 + (lane&15), khalf = lane>>4
//   element j corresponds to K = kc*32 + khalf*16 + j   (contiguous k order, same as A)
// => 16 contiguous f16 per lane; a wave's fragment load is one coalesced 1KB.
// ---------------------------------------------------------------------------
__global__ __launch_bounds__(256) void pack_w_kernel(const float* __restrict__ W,
                                                     _Float16* __restrict__ Wp) {
  int t    = blockIdx.x * 256 + threadIdx.x;   // 8192 threads total
  int lane = t & 31;
  int kc   = (t >> 5) & 31;
  int nt   = t >> 10;                          // 0..7
  int n    = nt * 16 + (lane & 15);
  int kh   = lane >> 4;
  const float* src = W + (size_t)n * L_DIM + kc * KCHUNK + kh * 16;
  _Float16* dst = Wp + (((size_t)nt * NKC + kc) * 32 + lane) * 16;
#pragma unroll
  for (int j = 0; j < 16; ++j) dst[j] = (_Float16)src[j];
}

// ---------------------------------------------------------------------------
// Main GEMM: grid = 256 WGs (B/128 row-blocks) x 256 threads (8 waves).
// Wave w owns N-tile w (16 cols) and all 128 rows of the block.
// A chunk (128 rows x 32 K) staged in double-buffered LDS as f16.
// A-fragment LDS loads are software-pipelined one mt ahead; the B fragment
// for K-chunk kc+1 is prefetched a whole iteration early.
// ---------------------------------------------------------------------------
__global__ __launch_bounds__(256) void branched_policy_gemm(
    const float* __restrict__ feat, const _Float16* __restrict__ Wp,
    const float* __restrict__ bias, float* __restrict__ out) {
  __shared__ _Float16 lds[2][MBLK * LSTR];     // 2 x 10KB

  const int tid  = threadIdx.x;
  const int wave = tid >> 5;                   // 0..7 == N-tile
  const int lane = tid & 31;
  const int l15  = lane & 15;
  const int kh   = lane >> 4;
  const int mblk = blockIdx.x;                 // 0..255

  const int   nGlob = wave * 16 + l15;
  const float bv    = bias[nGlob];

  v8f acc[8];
#pragma unroll
  for (int mt = 0; mt < 8; ++mt) {
#pragma unroll
    for (int r = 0; r < 8; ++r) acc[mt][r] = bv;   // seed with bias
  }

  // Staging role: thread t copies row (t>>1), half (t&1) -> 16 f32 per chunk.
  const int srow  = tid >> 1;
  const int shalf = tid & 1;
  const float4* aRow =
      (const float4*)(feat + ((size_t)mblk * MBLK + srow) * L_DIM + shalf * 16);

  // Per-lane packed-B base for this wave's N-tile (32B/lane/chunk, L2-hot).
  const _Float16* bp = Wp + (((size_t)wave * NKC) * 32 + lane) * 16;

  // ---- prefetch + stage chunk 0; prefetch B fragment for chunk 0 ----
  v16h bfrag = *(const v16h*)(bp);
  float4 p0 = aRow[0], p1 = aRow[1], p2 = aRow[2], p3 = aRow[3];
  {
    h8 lo = {(_Float16)p0.x, (_Float16)p0.y, (_Float16)p0.z, (_Float16)p0.w,
             (_Float16)p1.x, (_Float16)p1.y, (_Float16)p1.z, (_Float16)p1.w};
    h8 hi = {(_Float16)p2.x, (_Float16)p2.y, (_Float16)p2.z, (_Float16)p2.w,
             (_Float16)p3.x, (_Float16)p3.y, (_Float16)p3.z, (_Float16)p3.w};
    *(h8*)&lds[0][srow * LSTR + shalf * 16]     = lo;
    *(h8*)&lds[0][srow * LSTR + shalf * 16 + 8] = hi;
  }
  __syncthreads();

  for (int kc = 0; kc < NKC; ++kc) {
    const int cur = kc & 1;

    // Prefetch next A chunk (global) and next B fragment while WMMAs run.
    v16h bnext;
    if (kc + 1 < NKC) {
      const float4* ap = aRow + (size_t)(kc + 1) * 8;
      p0 = ap[0]; p1 = ap[1]; p2 = ap[2]; p3 = ap[3];
      bnext = *(const v16h*)(bp + (size_t)(kc + 1) * (32 * 16));
    }

    // A fragment, contiguous k order: lane(row=l15, kh) holds K run
    // [kh*16, kh*16+16) (matches packed B).
    const _Float16* abase = &lds[cur][l15 * LSTR + kh * 16];
    h8 c0 = *(const h8*)(abase);
    h8 c1 = *(const h8*)(abase + 8);

#pragma unroll
    for (int mt = 0; mt < 8; ++mt) {
      h8 n0, n1;
      if (mt + 1 < 8) {                        // issue next fragment's LDS
        const _Float16* nb = abase + (mt + 1) * 16 * LSTR;   // loads before
        n0 = *(const h8*)(nb);                 // this mt's WMMA
        n1 = *(const h8*)(nb + 8);
      }
      v16h afrag = __builtin_shufflevector(c0, c1, 0, 1, 2, 3, 4, 5, 6, 7,
                                           8, 9, 10, 11, 12, 13, 14, 15);
      acc[mt] = __builtin_amdgcn_wmma_f32_16x16x32_f16(
          false, afrag, false, bfrag, (short)0, acc[mt], false, false);
      if (mt + 1 < 8) { c0 = n0; c1 = n1; }
    }
    __syncthreads();                           // all waves done reading lds[cur]

    if (kc + 1 < NKC) {
      h8 lo = {(_Float16)p0.x, (_Float16)p0.y, (_Float16)p0.z, (_Float16)p0.w,
               (_Float16)p1.x, (_Float16)p1.y, (_Float16)p1.z, (_Float16)p1.w};
      h8 hi = {(_Float16)p2.x, (_Float16)p2.y, (_Float16)p2.z, (_Float16)p2.w,
               (_Float16)p3.x, (_Float16)p3.y, (_Float16)p3.z, (_Float16)p3.w};
      const int nxt = 1 - cur;
      *(h8*)&lds[nxt][srow * LSTR + shalf * 16]     = lo;
      *(h8*)&lds[nxt][srow * LSTR + shalf * 16 + 8] = hi;
      bfrag = bnext;
      __syncthreads();                         // staged chunk visible
    }
  }

  // ---- epilogue: tanh + scatter into the two output planes ----
  // C/D 16x16 f32 layout: lane(l15=n, kh) VGPR r -> row = kh*8 + r.
  const int    aIdx = nGlob >> 1;
  const int    sel  = nGlob & 1;
  float*       ob   = out + (size_t)sel * ((size_t)B_ROWS * A_BR) + aIdx;
  const int    rb   = mblk * MBLK + kh * 8;
#pragma unroll
  for (int mt = 0; mt < 8; ++mt) {
#pragma unroll
    for (int r = 0; r < 8; ++r) {
      const int row = rb + mt * 16 + r;
      ob[(size_t)row * A_BR] = tanhf(acc[mt][r]);
    }
  }
}

extern "C" void kernel_launch(void* const* d_in, const int* in_sizes, int n_in,
                              void* d_out, int out_size, void* d_ws, size_t ws_size,
                              hipStream_t stream) {
  (void)in_sizes; (void)n_in; (void)out_size; (void)ws_size;
  const float* feat = (const float*)d_in[0];   // [32768, 1024] f32
  const float* W    = (const float*)d_in[1];   // [64, 2, 1024] f32
  const float* bias = (const float*)d_in[2];   // [64, 2] f32
  float* out  = (float*)d_out;                 // out0 flat ++ out1 flat
  _Float16* Wp = (_Float16*)d_ws;              // 8*32*32*16 f16 = 256 KB scratch

  pack_w_kernel<<<32, 256, 0, stream>>>(W, Wp);
  branched_policy_gemm<<<256, 256, 0, stream>>>(feat, Wp, bias, out);
}
